// FLAGatedLinearAttention_5351529251273
// MI455X (gfx1250) — matrix operand order repaired and checked by reading, not compiled, not run on hardware
//
#include <hip/hip_runtime.h>
#include <hip/hip_bf16.h>

// ---------------------------------------------------------------------------
// FLA gated linear attention, MI455X (gfx1250, wave32, WMMA).
// B=2 T=4096 D=1024 H=16 KD=512 VD=1024 HK=32 HV=64 CHUNK=64 GATE_RANK=16
// ---------------------------------------------------------------------------

#define BT      8192      // B*T rows
#define DIM     1024
#define KD      512
#define VD      1024
#define NH      16
#define HK      32
#define HV      64
#define CHUNKI  64
#define NCHUNK  64
#define GR      16

#define AS1 __attribute__((address_space(1)))
#define AS3 __attribute__((address_space(3)))

#if defined(__gfx1250__) && __has_builtin(__builtin_amdgcn_global_load_async_to_lds_b128) && __has_builtin(__builtin_amdgcn_s_wait_asynccnt)
#define USE_ASYNC_LDS 1
#else
#define USE_ASYNC_LDS 0
#endif

typedef __attribute__((ext_vector_type(2)))  float  v2f;
typedef __attribute__((ext_vector_type(8)))  float  v8f;
typedef __attribute__((ext_vector_type(16))) __bf16 v16bf;
typedef __attribute__((__vector_size__(16))) int    v4i;   // builtin's b128 unit

#if USE_ASYNC_LDS
// async copy of 16B global -> LDS; src/dst cast to the builtin's exact types
#define GLD_ASYNC_B128(gsrc, ldst)                                          \
    __builtin_amdgcn_global_load_async_to_lds_b128(                         \
        (AS1 v4i*)(gsrc), (AS3 v4i*)(ldst), 0, 0)
#endif

union BF16Frag { v16bf v; uint4 q2[2]; unsigned short u[16]; };

static __device__ __forceinline__ unsigned short f32_to_bf16(float f) {
    unsigned int x = __float_as_uint(f);
    x = x + 0x7FFFu + ((x >> 16) & 1u);   // round to nearest even
    return (unsigned short)(x >> 16);
}
static __device__ __forceinline__ unsigned int pack_bf16(float lo, float hi) {
    return (unsigned int)f32_to_bf16(lo) | ((unsigned int)f32_to_bf16(hi) << 16);
}

// ---------------------------------------------------------------------------
// Kernel 1/5: C[M,N] = A[M,K] @ Bm[K,N], f32 in/out, bf16 WMMA, f32 accum.
// BLOCK 128x64, BLOCK_K 32, 8 waves, each a 32x32 tile (2x2 wmma frags).
// LDS layouts are fragment-native: each lane's operand = contiguous 16B runs
// (ds_load_b128), rows padded to 80B for bank-conflict-free access.
// M % 128 == 0, K % 32 == 0; N ragged (guarded), e.g. Wgk1 N=16.
// ---------------------------------------------------------------------------
#define GM 128
#define GN 64
#define GK 32
#define APAD 40   // ushorts per row: 32 data + 8 pad -> 80B row stride
#define BPAD 40

__global__ __launch_bounds__(256) void gemm_bf16wmma(
    const float* __restrict__ A, const float* __restrict__ Bm,
    float* __restrict__ C, int M, int N, int K)
{
    __shared__ alignas(16) unsigned short As[GM][APAD];   // [m][k] row-major
    __shared__ alignas(16) unsigned short Bt[GN][BPAD];   // [n][k] transposed

    const int tid  = threadIdx.x;
    const int wave = tid >> 5;
    const int lane = tid & 31;
    const int r    = lane & 15;      // col (B/C/D) or row (A) within frag
    const int cg   = lane >> 4;      // lane-half selector
    const int m0   = blockIdx.x * GM;
    const int n0   = blockIdx.y * GN;
    const int wm   = (wave >> 1) * 32;   // 0,32,64,96
    const int wn   = (wave & 1) * 32;    // 0,32

    v8f acc[2][2] = {};

    for (int k0 = 0; k0 < K; k0 += GK) {
        // stage A tile: pack 2 x f32->bf16 per u32 store
        for (int e = tid; e < GM * GK / 2; e += 256) {
            int rr = e >> 4, kp = e & 15;
            const float* src = A + (size_t)(m0 + rr) * K + k0 + 2 * kp;
            *(unsigned int*)&As[rr][2 * kp] = pack_bf16(src[0], src[1]);
        }
        // stage B tile transposed, N-guarded
        for (int e = tid; e < GN * GK / 2; e += 256) {
            int nn = e & 63, kp = e >> 6;
            int nc = n0 + nn;
            float lo = 0.0f, hi = 0.0f;
            if (nc < N) {
                lo = Bm[(size_t)(k0 + 2 * kp) * N + nc];
                hi = Bm[(size_t)(k0 + 2 * kp + 1) * N + nc];
            }
            *(unsigned int*)&Bt[nn][2 * kp] = pack_bf16(lo, hi);
        }
        if (k0 + GK < K) {   // prefetch next K tile -> global_prefetch_b8
            __builtin_prefetch(A + (size_t)(m0 + (tid >> 1)) * K + k0 + GK + (tid & 1) * 16, 0, 1);
            __builtin_prefetch(Bm + (size_t)(k0 + GK + (tid >> 6)) * N + n0 + (tid & 63), 0, 1);
        }
        __syncthreads();

        BF16Frag afr[2], bfr[2];
        // A 16x32 frag: lane(r,cg) needs K in [8cg,8cg+8) u [16+8cg,16+8cg+8)
        #pragma unroll
        for (int i = 0; i < 2; ++i) {
            int mrow = wm + i * 16 + r;
            afr[i].q2[0] = *(const uint4*)&As[mrow][cg * 8];
            afr[i].q2[1] = *(const uint4*)&As[mrow][16 + cg * 8];
        }
        // B 32x16 frag: lane(col=r,cg) needs K in [8cg,8cg+8) u [16+8cg,16+8cg+8)
        #pragma unroll
        for (int j = 0; j < 2; ++j) {
            int nc = wn + j * 16 + r;
            bfr[j].q2[0] = *(const uint4*)&Bt[nc][cg * 8];
            bfr[j].q2[1] = *(const uint4*)&Bt[nc][16 + cg * 8];
        }
        #pragma unroll
        for (int i = 0; i < 2; ++i)
            #pragma unroll
            for (int j = 0; j < 2; ++j)
                acc[i][j] = __builtin_amdgcn_wmma_f32_16x16x32_bf16(
                    false, afr[i].v, false, bfr[j].v,
                    (short)0, acc[i][j], false, false);
        __syncthreads();
    }

    // D layout: col = r, row = 16*i + 8*cg + idx
    for (int i = 0; i < 2; ++i)
        for (int j = 0; j < 2; ++j) {
            int col = n0 + wn + j * 16 + r;
            if (col >= N) continue;
            #pragma unroll
            for (int idx = 0; idx < 8; ++idx) {
                int row = m0 + wm + i * 16 + cg * 8 + idx;
                C[(size_t)row * N + col] = acc[i][j][idx];
            }
        }
}

// ---------------------------------------------------------------------------
// Kernel 2: gk = log_sigmoid(xg1 @ Wgk2 + bgk) / 16   (rank-16, VALU)
// ---------------------------------------------------------------------------
__global__ __launch_bounds__(256) void gate_kernel(
    const float* __restrict__ xg1, const float* __restrict__ Wgk2,
    const float* __restrict__ bgk, float* __restrict__ gk)
{
    __shared__ float xs[GR];
    const int row = blockIdx.x;
    if (threadIdx.x < GR) xs[threadIdx.x] = xg1[(size_t)row * GR + threadIdx.x];
    __syncthreads();
    for (int col = threadIdx.x; col < KD; col += 256) {
        float z = bgk[col];
        #pragma unroll
        for (int i = 0; i < GR; ++i) z += xs[i] * Wgk2[i * KD + col];
        float ls = fminf(z, 0.0f) - log1pf(__expf(-fabsf(z)));  // log_sigmoid
        gk[(size_t)row * KD + col] = ls * (1.0f / 16.0f);       // / GATE_NORM
    }
}

// ---------------------------------------------------------------------------
// Kernel 3: chunked gated linear attention scan. One workgroup per (b,h);
// state S (stored transposed, ST[v][k]) lives in LDS for the whole T.
// q/k/gk tiles arrive via GLOBAL_LOAD_ASYNC_TO_LDS_B128 (ASYNCcnt) when the
// toolchain declares the builtin. All GEMMs use v_wmma_f32_16x16x4_f32.
//   A  = (q*e^b*s) @ (k*e^-b)^T, tril-masked        (64x64, K=32)
//   o  = A @ v + qe @ S                              (64x64, K=64+32)
//   S  = e^{b_last} * (S + (k*e^-b)^T @ v)           (32x64, K=64)
// ---------------------------------------------------------------------------
#define VCP (CHUNKI + 2)   // 66: 8B-aligned pairs, low-conflict
#define STP (HK + 2)       // 34

__global__ __launch_bounds__(256) void fla_scan_kernel(
    const float* __restrict__ qb, const float* __restrict__ kb,
    const float* __restrict__ vb, const float* __restrict__ gkb,
    float* __restrict__ ob)
{
    __shared__ alignas(16) float s_qe[CHUNKI * HK];       // qe[t][f]
    __shared__ alignas(16) float s_kn[CHUNKI * HK];       // k*e^-b [t][f]
    __shared__ alignas(16) float s_vcT[HV * VCP];         // v transposed [u][t]
    __shared__ alignas(16) float s_ST[HV * STP];          // S transposed [u][f]
    __shared__ alignas(16) float s_scr[CHUNKI * CHUNKI];  // phase1: b; phase2: A
    __shared__ float s_blast[HK];

    const int tid  = threadIdx.x;
    const int wave = tid >> 5;
    const int lane = tid & 31;
    const int r    = lane & 15;
    const int cg   = lane >> 4;
    const int bh   = blockIdx.x;          // 0..31
    const int bidx = bh >> 4;
    const int h    = bh & 15;
    const float scale = 0.17677669529663687f;   // HK^-0.5

    const float* qh = qb  + (size_t)bidx * 4096 * KD + h * HK;
    const float* kh = kb  + (size_t)bidx * 4096 * KD + h * HK;
    const float* gh = gkb + (size_t)bidx * 4096 * KD + h * HK;
    const float* vh = vb  + (size_t)bidx * 4096 * VD + h * HV;
    float*       oh = ob  + (size_t)bidx * 4096 * VD + h * HV;

    for (int i = tid; i < HV * STP; i += 256) s_ST[i] = 0.0f;
    __syncthreads();

    for (int cidx = 0; cidx < NCHUNK; ++cidx) {
        const size_t t0 = (size_t)cidx * CHUNKI;

        // ---- stage q,k,gk (64x32) and v (64x64, transposed) ----
#if USE_ASYNC_LDS
        for (int e = tid; e < (CHUNKI * HK) / 4; e += 256) {   // 512 b128 ops
            int t = e >> 3, fq = (e & 7) * 4;
            size_t goff = (t0 + t) * (size_t)KD + fq;
            GLD_ASYNC_B128(qh + goff, s_qe  + t * HK + fq);
            GLD_ASYNC_B128(kh + goff, s_kn  + t * HK + fq);
            GLD_ASYNC_B128(gh + goff, s_scr + t * HK + fq);
        }
#else
        for (int e = tid; e < CHUNKI * HK; e += 256) {
            int t = e >> 5, f = e & 31;
            s_qe[e]  = qh[(t0 + t) * KD + f];
            s_kn[e]  = kh[(t0 + t) * KD + f];
            s_scr[e] = gh[(t0 + t) * KD + f];
        }
#endif
        for (int e = tid; e < CHUNKI * HV; e += 256) {
            int t = e >> 6, u = e & 63;
            s_vcT[u * VCP + t] = vh[(t0 + t) * VD + u];
        }
#if USE_ASYNC_LDS
        __builtin_amdgcn_s_wait_asynccnt(0);
#endif
        __syncthreads();

        // ---- cumsum over time per feature ----
        if (tid < HK) {
            float a = 0.0f;
            for (int t = 0; t < CHUNKI; ++t) {
                a += s_scr[t * HK + tid];
                s_scr[t * HK + tid] = a;
            }
            s_blast[tid] = a;
        }
        __syncthreads();

        // ---- qe = q*exp(b)*scale ; kneg = k*exp(-b) ----
        for (int e = tid; e < CHUNKI * HK; e += 256) {
            float bv = s_scr[e];
            s_qe[e] = s_qe[e] * __expf(bv) * scale;
            s_kn[e] = s_kn[e] * __expf(-bv);
        }
        __syncthreads();

        // ---- A = qe @ kneg^T, masked, into s_scr (2 tiles/wave) ----
        {
            const int ti = wave >> 1;
            for (int jj = 0; jj < 2; ++jj) {
                const int tj = 2 * (wave & 1) + jj;
                v8f acc = {};
                if (tj <= ti) {
                    #pragma unroll
                    for (int kk = 0; kk < 8; ++kk) {
                        const int k0 = kk * 4 + cg * 2;
                        v2f af, bf;
                        af[0] = s_qe[(ti * 16 + r) * HK + k0];
                        af[1] = s_qe[(ti * 16 + r) * HK + k0 + 1];
                        bf[0] = s_kn[(tj * 16 + r) * HK + k0];
                        bf[1] = s_kn[(tj * 16 + r) * HK + k0 + 1];
                        acc = __builtin_amdgcn_wmma_f32_16x16x4_f32(
                            false, af, false, bf, (short)0, acc, false, false);
                    }
                }
                #pragma unroll
                for (int idx = 0; idx < 8; ++idx) {
                    int ig = ti * 16 + cg * 8 + idx;
                    int jg = tj * 16 + r;
                    s_scr[ig * CHUNKI + jg] = (jg <= ig) ? acc[idx] : 0.0f;
                }
            }
        }
        __syncthreads();

        // ---- o = A @ v + qe @ S  (2 tiles/wave), store to global ----
        {
            const int ti = wave >> 1;
            for (int jj = 0; jj < 2; ++jj) {
                const int tv = 2 * (wave & 1) + jj;
                const int vcb = (tv * 16 + r) * VCP;   // lane's v column base
                const int stb = (tv * 16 + r) * STP;   // lane's S column base
                v8f acc = {};
                #pragma unroll
                for (int kk = 0; kk < 16; ++kk) {           // A @ v, K=64
                    const int k0 = kk * 4 + cg * 2;
                    v2f af, bf;
                    af[0] = s_scr[(ti * 16 + r) * CHUNKI + k0];
                    af[1] = s_scr[(ti * 16 + r) * CHUNKI + k0 + 1];
                    bf[0] = s_vcT[vcb + k0];
                    bf[1] = s_vcT[vcb + k0 + 1];
                    acc = __builtin_amdgcn_wmma_f32_16x16x4_f32(
                        false, af, false, bf, (short)0, acc, false, false);
                }
                #pragma unroll
                for (int kk = 0; kk < 8; ++kk) {            // qe @ S, K=32
                    const int k0 = kk * 4 + cg * 2;
                    v2f af, bf;
                    af[0] = s_qe[(ti * 16 + r) * HK + k0];
                    af[1] = s_qe[(ti * 16 + r) * HK + k0 + 1];
                    bf[0] = s_ST[stb + k0];
                    bf[1] = s_ST[stb + k0 + 1];
                    acc = __builtin_amdgcn_wmma_f32_16x16x4_f32(
                        false, af, false, bf, (short)0, acc, false, false);
                }
                #pragma unroll
                for (int idx = 0; idx < 8; ++idx) {
                    int ig = ti * 16 + cg * 8 + idx;
                    int vg = tv * 16 + r;
                    oh[(t0 + ig) * VD + vg] = acc[idx];
                }
            }
        }
        __syncthreads();   // all qe@S reads of s_ST complete

        // ---- S = exp(b_last) * (S + kneg^T @ v)  (1 tile/wave) ----
        {
            const int tm = wave >> 2;       // 0..1 over HK
            const int tv = wave & 3;        // 0..3 over HV
            const int vcb = (tv * 16 + r) * VCP;
            const int stb = (tv * 16 + r) * STP;
            v8f acc;
            #pragma unroll
            for (int idx = 0; idx < 8; ++idx)
                acc[idx] = s_ST[stb + tm * 16 + cg * 8 + idx];
            #pragma unroll
            for (int kk = 0; kk < 16; ++kk) {               // K = 64 (time)
                const int k0 = kk * 4 + cg * 2;
                v2f af, bf;
                af[0] = s_kn[(k0)     * HK + tm * 16 + r];  // kneg^T
                af[1] = s_kn[(k0 + 1) * HK + tm * 16 + r];
                bf[0] = s_vcT[vcb + k0];
                bf[1] = s_vcT[vcb + k0 + 1];
                acc = __builtin_amdgcn_wmma_f32_16x16x4_f32(
                    false, af, false, bf, (short)0, acc, false, false);
            }
            #pragma unroll
            for (int idx = 0; idx < 8; ++idx) {
                int mg = tm * 16 + cg * 8 + idx;
                s_ST[stb + mg] = __expf(s_blast[mg]) * acc[idx];
            }
        }
        __syncthreads();
    }
}

// ---------------------------------------------------------------------------
// Kernel 4: RMS-norm over HV + norm_w + SiLU(g), in place on o.
// ---------------------------------------------------------------------------
__global__ __launch_bounds__(256) void norm_silu_kernel(
    const float* __restrict__ gbuf, const float* __restrict__ norm_w,
    float* __restrict__ obuf)
{
    const int unit = blockIdx.x * 8 + (threadIdx.x >> 5);
    const int lane = threadIdx.x & 31;
    const size_t base = (size_t)(unit >> 4) * VD + (size_t)(unit & 15) * HV;

    float o0 = obuf[base + lane], o1 = obuf[base + 32 + lane];
    float s = o0 * o0 + o1 * o1;
    #pragma unroll
    for (int off = 16; off; off >>= 1) s += __shfl_xor(s, off, 32);
    const float rinv = rsqrtf(s * (1.0f / 64.0f) + 1e-5f);

    float g0 = gbuf[base + lane], g1 = gbuf[base + 32 + lane];
    obuf[base + lane]      = o0 * rinv * norm_w[lane]      * (g0 / (1.0f + __expf(-g0)));
    obuf[base + 32 + lane] = o1 * rinv * norm_w[lane + 32] * (g1 / (1.0f + __expf(-g1)));
}

// ---------------------------------------------------------------------------
// Launch. Workspace layout (floats):
//   q 4.19M | k 4.19M | v 8.39M | g 8.39M | gk 4.19M | xg1 131K | o 8.39M
//   total 37,879,808 floats (~151.5 MB)
// ---------------------------------------------------------------------------
extern "C" void kernel_launch(void* const* d_in, const int* in_sizes, int n_in,
                              void* d_out, int out_size, void* d_ws, size_t ws_size,
                              hipStream_t stream) {
    const float* x      = (const float*)d_in[0];
    const float* Wq     = (const float*)d_in[1];
    const float* Wk     = (const float*)d_in[2];
    const float* Wv     = (const float*)d_in[3];
    const float* Wgk1   = (const float*)d_in[4];
    const float* Wgk2   = (const float*)d_in[5];
    const float* bgk    = (const float*)d_in[6];
    const float* Wg     = (const float*)d_in[7];
    const float* norm_w = (const float*)d_in[8];
    const float* Wo     = (const float*)d_in[9];
    float* out = (float*)d_out;

    float* ws  = (float*)d_ws;
    float* q   = ws;
    float* k   = q   + (size_t)BT * KD;
    float* v   = k   + (size_t)BT * KD;
    float* g   = v   + (size_t)BT * VD;
    float* gk  = g   + (size_t)BT * VD;
    float* xg1 = gk  + (size_t)BT * KD;
    float* o   = xg1 + (size_t)BT * GR;

    const dim3 blk(256);
    gemm_bf16wmma<<<dim3(BT / GM,  KD / GN), blk, 0, stream>>>(x, Wq,   q,   BT, KD,  DIM);
    gemm_bf16wmma<<<dim3(BT / GM,  KD / GN), blk, 0, stream>>>(x, Wk,   k,   BT, KD,  DIM);
    gemm_bf16wmma<<<dim3(BT / GM,  VD / GN), blk, 0, stream>>>(x, Wv,   v,   BT, VD,  DIM);
    gemm_bf16wmma<<<dim3(BT / GM,  VD / GN), blk, 0, stream>>>(x, Wg,   g,   BT, VD,  DIM);
    gemm_bf16wmma<<<dim3(BT / GM,  1),       blk, 0, stream>>>(x, Wgk1, xg1, BT, GR,  DIM);

    gate_kernel<<<BT, blk, 0, stream>>>(xg1, Wgk2, bgk, gk);

    fla_scan_kernel<<<2 * NH, blk, 0, stream>>>(q, k, v, gk, o);

    norm_silu_kernel<<<(BT * NH) / 8, blk, 0, stream>>>(g, norm_w, o);

    gemm_bf16wmma<<<dim3(BT / GM, DIM / GN), blk, 0, stream>>>(o, Wo, out, BT, DIM, DIM);
}
